// LSTM_12249246728290
// MI455X (gfx1250) — compile-verified, matrix operand order checked by reading, not run on hardware
//
#include <hip/hip_runtime.h>
#include <hip/hip_bf16.h>
#include <stdint.h>

typedef __attribute__((ext_vector_type(16))) __bf16 v16bf;
typedef __attribute__((ext_vector_type(8)))  float  v8f;

#define T_STEPS 8192
#define IN_DIM  1024
#define HID     1024
#define G4      4096          // 4*HID gate columns
#define NWG     64            // persistent workgroups in scan
#define HPW     16            // hidden units per workgroup
#define WLDS_STRIDE 513       // 512 pairs + 1 pad dword -> conflict-free banks

// ---- workspace byte offsets (total ~160 MB) ----
#define WS_SYNC   0u
#define WS_HBUF   256u                      // 2*1024*4 = 8192 B  (h ping-pong)
#define WS_XB     8448u                     // 8192*1024*2 = 16777216 B (x in bf16)
#define WS_WXT    16785664u                 // 4096*1024*2 = 8388608 B (Wx^T bf16, [n][k])
#define WS_WHT    25174272u                 // 8388608 B (Wh^T bf16, [n][k])
#define WS_ZX     33562880u                 // 8192*4096*4 = 134217728 B (X@Wx + b, f32)

static __device__ __forceinline__ uint16_t f32_to_bf16(float f) {
  uint32_t u = __float_as_uint(f);
  uint32_t r = u + 0x7fffu + ((u >> 16) & 1u);   // round-to-nearest-even
  return (uint16_t)(r >> 16);
}
static __device__ __forceinline__ float sigmoidf(float x) {
  return 1.0f / (1.0f + __expf(-x));
}
// bf16 -> f32 through the real __bf16 type so the backend can select
// v_cvt_f32_bf16 (OPSEL for hi half) / fuse into v_fma_mix_f32_bf16.
static __device__ __forceinline__ float bf16_bits_to_f32(uint16_t bits) {
  union { uint16_t u; __bf16 b; } cv{bits};
  return (float)cv.b;
}

// ---------------- phase -1: reset cross-launch scratch state ----------------
__global__ void k_init(unsigned* __restrict__ syncc, float* __restrict__ hbuf) {
  if (threadIdx.x == 0) *syncc = 0u;
  for (int i = threadIdx.x; i < 2 * HID; i += blockDim.x) hbuf[i] = 0.0f;
}

// ---------------- phase 0a: x (f32) -> x_bf16 ----------------
__global__ void k_cvt_x(const float* __restrict__ x, uint16_t* __restrict__ xb) {
  int i = blockIdx.x * blockDim.x + threadIdx.x;   // 8M elements
  xb[i] = f32_to_bf16(x[i]);
}

// ---------------- phase 0b: W (f32 [2048][4096]) -> WxT / WhT (bf16 [4096][1024]) ----
__global__ void k_cvt_w(const float* __restrict__ W,
                        uint16_t* __restrict__ wxt, uint16_t* __restrict__ wht) {
  int i = blockIdx.x * blockDim.x + threadIdx.x;   // 2 * 4096 * 1024 elements
  int part = i >> 22;                              // 0 = Wx rows, 1 = Wh rows
  int r = i & ((1 << 22) - 1);
  int n = r >> 10;
  int k = r & (IN_DIM - 1);
  float v = W[(size_t)(part * IN_DIM + k) * G4 + n];
  uint16_t* dst = part ? wht : wxt;
  dst[(size_t)n * IN_DIM + k] = f32_to_bf16(v);    // k-contiguous writes: coalesced
}

// ---------------- phase 1: Zx = X @ Wx + b  (WMMA bf16, f32 accum) ----------------
// 64x64 output tile per wave: 4 A-frags x 4 B-frags -> 16 WMMAs per K=32 step.
// 4x fewer fragment bytes per WMMA than a 16x16 tile -> near the L2-traffic floor.
__global__ __launch_bounds__(256) void k_gemm_zx(
    const uint32_t* __restrict__ xb,   // bf16 pairs, [8192][512]
    const uint32_t* __restrict__ wxt,  // bf16 pairs, [4096][512]  (B^T: row n, k-contig)
    const float*    __restrict__ bias, // [4096]
    float*          __restrict__ zx) { // [8192][4096]
  const int lane = threadIdx.x & 31;
  const int wave = threadIdx.x >> 5;
  // block covers 128(t) x 256(n); wave (tw, nw) covers 64x64
  const int t0 = blockIdx.y * 128 + (wave & 1) * 64;
  const int n0 = blockIdx.x * 256 + (wave >> 1) * 64;
  const int mr = lane & 15;       // A row / B column within a 16-tile
  const int kh = lane >> 4;       // K-half select

  const uint32_t* arow[4];
  const uint32_t* brow[4];
#pragma unroll
  for (int i = 0; i < 4; ++i) {
    arow[i] = xb  + (size_t)(t0 + i * 16 + mr) * (IN_DIM / 2);
    brow[i] = wxt + (size_t)(n0 + i * 16 + mr) * (IN_DIM / 2);
  }

  v8f acc[4][4];
#pragma unroll
  for (int a = 0; a < 4; ++a)
#pragma unroll
    for (int bn = 0; bn < 4; ++bn)
      acc[a][bn] = v8f{0.f, 0.f, 0.f, 0.f, 0.f, 0.f, 0.f, 0.f};

  for (int kk = 0; kk < IN_DIM; kk += 32) {
    union { v16bf v; uint32_t u[8]; } A[4], B[4];
    const int base = (kk >> 1) + kh * 4;  // pair index: k/2
#pragma unroll
    for (int i = 0; i < 4; ++i) {
#pragma unroll
      for (int v = 0; v < 8; ++v) {
        // VGPR v holds K pair = kk + (v>=4?16:0) + kh*8 + (v&3)*2
        const int p = base + ((v & 4) ? 8 : 0) + (v & 3);
        A[i].u[v] = arow[i][p];
        B[i].u[v] = brow[i][p];
      }
    }
#pragma unroll
    for (int a = 0; a < 4; ++a)
#pragma unroll
      for (int bn = 0; bn < 4; ++bn)
        acc[a][bn] = __builtin_amdgcn_wmma_f32_16x16x32_bf16(
            false, A[a].v, false, B[bn].v, (short)0, acc[a][bn], false, false);
  }

#pragma unroll
  for (int bn = 0; bn < 4; ++bn) {
    const int n = n0 + bn * 16 + (lane & 15);
    const float bv = bias[n];
#pragma unroll
    for (int a = 0; a < 4; ++a) {
      const int mrow = t0 + a * 16 + ((lane >> 4) << 3);   // +8 for lanes 16..31
#pragma unroll
      for (int r = 0; r < 8; ++r)
        zx[(size_t)(mrow + r) * G4 + n] = acc[a][bn][r] + bv;
    }
  }
}

// ---------------- phase 2: persistent sequential LSTM scan ----------------
// 64 workgroups x 256 threads. WG wg owns hidden units [wg*16, wg*16+16) and the
// 64 gate columns {g*1024 + unit}. Recurrent weights live in LDS for all 8192 steps.
__global__ __launch_bounds__(256) void k_lstm_scan(
    const float*    __restrict__ zx,    // [T][4096] input projection (+bias)
    const uint32_t* __restrict__ wht,   // bf16 pairs, [4096][512]
    float*          __restrict__ out,   // [T][1024]
    float*          __restrict__ hbuf,  // [2][1024] ping-pong
    unsigned*       __restrict__ syncc) {
  __shared__ uint32_t wlds[64 * WLDS_STRIDE];   // 131,328 B: 64 cols x 512 bf16-pairs (padded)
  __shared__ float    hlds[HID];                // 4 KB
  __shared__ float    part[4][64];
  __shared__ float    zsum[64];

  const int tid = threadIdx.x;
  const int wg  = blockIdx.x;
  const int h0  = wg * HPW;

  // one-time: recurrent weight slice -> LDS
  for (int idx = tid; idx < 64 * 512; idx += 256) {
    const int c = idx >> 9;            // local column 0..63  (c = gate*16 + unit)
    const int p = idx & 511;           // k pair 0..511
    const int n = (c >> 4) * HID + h0 + (c & 15);
    wlds[c * WLDS_STRIDE + p] = wht[(size_t)n * (IN_DIM / 2) + p];
  }
  __syncthreads();

  const int q = tid >> 6;              // K quarter 0..3 (uniform within a wave)
  const int c = tid & 63;              // local column  (lane-varying -> bank-conflict-free)
  const uint32_t* wrow = &wlds[c * WLDS_STRIDE + q * 128];

  float cst = 0.0f;                    // cell state for unit `tid` (threads 0..15)

  for (int t = 0; t < T_STEPS; ++t) {
    // stage h_{t-1} into LDS (broadcast-read later)
    const float* hsrc = hbuf + (t & 1) * HID;
#pragma unroll
    for (int r = 0; r < 4; ++r) hlds[tid + r * 256] = hsrc[tid + r * 256];
    __syncthreads();

    // warm L2/L0 for next step's input-projection row while we do the dot
    if (t + 1 < T_STEPS && tid < 64) {
      const int n = (tid >> 4) * HID + h0 + (tid & 15);
      __builtin_prefetch(&zx[(size_t)(t + 1) * G4 + n], 0, 3);
    }

    // partial dot: 256 MACs per thread, weights from LDS, h broadcast from LDS
    float acc = 0.0f;
    const float2* h2 = ((const float2*)hlds) + q * 128;
#pragma unroll 16
    for (int p = 0; p < 128; ++p) {
      const uint32_t wp = wrow[p];
      const float2 hh = h2[p];
      acc = fmaf(bf16_bits_to_f32((uint16_t)wp), hh.x, acc);
      acc = fmaf(bf16_bits_to_f32((uint16_t)(wp >> 16)), hh.y, acc);
    }
    part[q][c] = acc;
    __syncthreads();

    if (tid < 64) {  // reduce K quarters, add input projection (bias already in zx)
      const int n = (tid >> 4) * HID + h0 + (tid & 15);
      zsum[tid] = part[0][tid] + part[1][tid] + part[2][tid] + part[3][tid]
                + zx[(size_t)t * G4 + n];
    }
    __syncthreads();

    if (tid < HPW) {  // gate math for this WG's 16 hidden units; c-state in registers
      const float zi = zsum[tid];
      const float zj = zsum[16 + tid];
      const float zf = zsum[32 + tid];
      const float zo = zsum[48 + tid];
      const float nc = cst * sigmoidf(zf + 1.0f) + sigmoidf(zi) * tanhf(zj);
      const float nh = tanhf(nc) * sigmoidf(zo);
      cst = nc;
      out[(size_t)t * HID + h0 + tid] = nh;
      hbuf[((t + 1) & 1) * HID + h0 + tid] = nh;
      __threadfence();
    }
    __syncthreads();

    // grid-wide step barrier: monotonic counter, no resets -> race-free
    if (tid == 0) {
      atomicAdd(syncc, 1u);
      const unsigned target = (unsigned)(t + 1) * NWG;
      while (__hip_atomic_load(syncc, __ATOMIC_ACQUIRE, __HIP_MEMORY_SCOPE_AGENT) < target)
        __builtin_amdgcn_s_sleep(1);
    }
    __syncthreads();
  }
}

extern "C" void kernel_launch(void* const* d_in, const int* in_sizes, int n_in,
                              void* d_out, int out_size, void* d_ws, size_t ws_size,
                              hipStream_t stream) {
  const float* x = (const float*)d_in[0];   // [8192][1024]
  const float* W = (const float*)d_in[1];   // [2048][4096]
  const float* b = (const float*)d_in[2];   // [4096]
  float* out = (float*)d_out;               // [8192][1024]

  char* ws = (char*)d_ws;
  unsigned*  syncc = (unsigned*)(ws + WS_SYNC);
  float*     hbuf  = (float*)(ws + WS_HBUF);
  uint16_t*  xb    = (uint16_t*)(ws + WS_XB);
  uint16_t*  wxt   = (uint16_t*)(ws + WS_WXT);
  uint16_t*  wht   = (uint16_t*)(ws + WS_WHT);
  float*     zx    = (float*)(ws + WS_ZX);

  // reset cross-launch state (ws is poisoned / left dirty between calls)
  k_init<<<1, 256, 0, stream>>>(syncc, hbuf);

  // phase 0: precision/layout conversion
  k_cvt_x<<<(T_STEPS * IN_DIM) / 256, 256, 0, stream>>>(x, xb);
  k_cvt_w<<<(2 * G4 * IN_DIM) / 256, 256, 0, stream>>>(W, wxt, wht);

  // phase 1: Zx = X @ Wx + b   (WMMA bf16 -> f32), 64x64 tile per wave
  k_gemm_zx<<<dim3(G4 / 256, T_STEPS / 128), 256, 0, stream>>>(
      (const uint32_t*)xb, (const uint32_t*)wxt, b, zx);

  // phase 2: sequential recurrence, weights LDS-resident, grid-synced per step
  k_lstm_scan<<<NWG, 256, 0, stream>>>(zx, (const uint32_t*)wht, out, hbuf, syncc);
}
